// FeatureAlignModule_16389595202200
// MI455X (gfx1250) — matrix operand order repaired and checked by reading, not compiled, not run on hardware
//
#include <hip/hip_runtime.h>

typedef __attribute__((ext_vector_type(16))) __bf16 v16bf;
typedef __attribute__((ext_vector_type(8)))  float  v8f;
typedef __attribute__((ext_vector_type(8)))  unsigned short ushort8;

// Problem sizes (fixed by the reference)
constexpr int B_   = 16;
constexpr int C_   = 144;          // feat channels (also offset-conv out ch)
constexpr int H_   = 64, W_ = 64;
constexpr int HW   = H_ * W_;      // 4096
constexpr int NPIX = B_ * HW;      // 65536
constexpr int CIN1 = 288;          // concat channels
constexpr int M1   = 144;          // offset conv out channels
constexpr int K1D  = CIN1 * 9;     // 2592 (mult of 32)
constexpr int M2   = 256;          // dcn out channels
constexpr int K2D  = C_ * 9;       // 1296
constexpr int K2P  = 1312;         // padded to multiple of 32
constexpr int HP   = 66, WP = 66;  // padded spatial (1-px halo)
constexpr int OFFG = 8;

// GEMM tiling
constexpr int TN   = 512;          // N per block (8 waves x 64)
constexpr int BS_S = 40;           // Bs row stride in bf16 (80B: 16B-aligned cells,
                                   // bank step 20 dwords -> conflict-free reads)
constexpr int AS_S = 34;           // As row stride (bank step 17 -> conflict-free)

__device__ __forceinline__ unsigned short f2bf(float f) {
  unsigned int u = __builtin_bit_cast(unsigned int, f);
  u += 0x7FFFu + ((u >> 16) & 1u);   // round-to-nearest-even
  return (unsigned short)(u >> 16);
}

// ---------------- K0a: pack concat input into padded bf16 tensor ----------
__global__ __launch_bounds__(256) void k_pack_xpad(
    const float* __restrict__ fa, const float* __restrict__ fu,
    unsigned short* __restrict__ xpad, int n) {
  int i = blockIdx.x * 256 + threadIdx.x;
  if (i >= n) return;
  int xx = i % WP;  int t = i / WP;
  int yy = t % HP;  t /= HP;
  int c  = t % CIN1;
  int b  = t / CIN1;
  float v = 0.f;
  if (yy >= 1 && yy <= H_ && xx >= 1 && xx <= W_) {
    const float* src = (c < C_) ? fa : fu;
    int cc = (c < C_) ? c : c - C_;
    v = src[((b * C_ + cc) * H_ + (yy - 1)) * W_ + (xx - 1)];
  }
  xpad[i] = f2bf(v);
}

// ---------------- K0b: off_w -> bf16 (natural [144][2592] flatten) --------
__global__ __launch_bounds__(256) void k_cvt_w1(
    const float* __restrict__ w, unsigned short* __restrict__ o, int n) {
  int i = blockIdx.x * 256 + threadIdx.x;
  if (i < n) o[i] = f2bf(w[i]);
}

// ---------------- K0c: dcn_w -> bf16 padded [256][1312] -------------------
__global__ __launch_bounds__(256) void k_cvt_w2(
    const float* __restrict__ w, unsigned short* __restrict__ o, int n) {
  int i = blockIdx.x * 256 + threadIdx.x;
  if (i >= n) return;
  int k = i % K2P, m = i / K2P;
  o[i] = (k < K2D) ? f2bf(w[m * K2D + k]) : (unsigned short)0;
}

// ---------------- K0d: zero pad rows of cols ------------------------------
__global__ __launch_bounds__(256) void k_zero_colpad(
    unsigned short* __restrict__ cols, int n) {
  int i = blockIdx.x * 256 + threadIdx.x;
  if (i < n) cols[K2D * NPIX + i] = 0;
}

// ---- shared wave-level fragment math: 16M x 64N per wave, 4 WMMAs/chunk --
// As: [16][AS_S] (k-contiguous pairs).  Bs: [TN][BS_S] (k-contiguous pairs).
// A frag pair offsets: ka = (v&3)*2 + (v>>2)*16 + half*8  (ISA A 16x32 layout)
// B frag pair offsets: kb = (v&3)*2 + (v>>2)*16 + half*8  (same k order as A)
__device__ __forceinline__ void wave_mma4(
    const unsigned short* As, const unsigned short* Bs,
    int wave, int half, int l15, v8f acc[4]) {
  union { v16bf v; unsigned int u[8]; } af;
#pragma unroll
  for (int v = 0; v < 8; ++v) {
    int ka = (v & 3) * 2 + (v >> 2) * 16 + half * 8;
    af.u[v] = *(const unsigned int*)&As[l15 * AS_S + ka];
  }
#pragma unroll
  for (int s = 0; s < 4; ++s) {
    union { v16bf v; unsigned int u[8]; } bfr;
    int nrow = wave * 64 + s * 16 + l15;
#pragma unroll
    for (int v = 0; v < 8; ++v) {
      int kb = (v & 3) * 2 + (v >> 2) * 16 + half * 8;
      bfr.u[v] = *(const unsigned int*)&Bs[nrow * BS_S + kb];
    }
    acc[s] = __builtin_amdgcn_wmma_f32_16x16x32_bf16(
        false, af.v, false, bfr.v, (short)0, acc[s], false, false);
  }
}

// ---------------- K1: offset conv as implicit GEMM (WMMA bf16) ------------
// Block: 256 threads (8 wave32 waves). Tile: 16 M x 512 N. K chunk = 32.
__global__ __launch_bounds__(256) void k_conv1_wmma(
    const unsigned short* __restrict__ xpad,
    const unsigned short* __restrict__ w1,
    const float* __restrict__ bias,
    float* __restrict__ offs) {
  __shared__ __align__(16) unsigned short As[16 * AS_S];
  __shared__ __align__(16) unsigned short Bs[TN * BS_S];
  const int tid  = threadIdx.x;
  const int wave = tid >> 5;
  const int lane = tid & 31;
  const int half = lane >> 4;
  const int l15  = lane & 15;
  const int n0 = blockIdx.x * TN;
  const int m0 = blockIdx.y * 16;
  v8f acc[4] = {{}, {}, {}, {}};
  for (int kc = 0; kc < K1D / 32; ++kc) {
    {  // stage A: 16x32 = 256 u32, one per thread
      int m = tid >> 4, kp = (tid & 15) * 2;
      *(unsigned int*)&As[m * AS_S + kp] =
          *(const unsigned int*)&w1[(m0 + m) * K1D + kc * 32 + kp];
    }
    // stage B (im2col gather): 512n x 32k; thread owns (n, 8 consecutive k)
#pragma unroll
    for (int i = 0; i < 8; ++i) {
      int idx = tid + i * 256;                // 0..2047
      int nl  = idx & 511;                    // lanes -> consecutive n
      int kg  = idx >> 9;                     // k-group of 8
      int n   = n0 + nl;
      int b   = n >> 12, pix = n & 4095;
      int y   = pix >> 6, x = pix & 63;
      ushort8 vals;
#pragma unroll
      for (int j = 0; j < 8; ++j) {
        int k  = kc * 32 + kg * 8 + j;
        int ci = k / 9;
        int r  = k - ci * 9;
        vals[j] = xpad[((b * CIN1 + ci) * HP + y + r / 3) * WP + x + r % 3];
      }
      *(ushort8*)&Bs[nl * BS_S + kg * 8] = vals;   // one ds_store_b128
    }
    __syncthreads();
    wave_mma4(As, Bs, wave, half, l15, acc);
    __syncthreads();
  }
#pragma unroll
  for (int s = 0; s < 4; ++s) {
    int n = n0 + wave * 64 + s * 16 + l15;
    int b = n >> 12, pix = n & 4095;
#pragma unroll
    for (int r = 0; r < 8; ++r) {
      int m = m0 + r + half * 8;
      offs[(b * M1 + m) * HW + pix] = acc[s][r] + bias[m];
    }
  }
}

// ---------------- K2: deformable bilinear sampling -> cols bf16 -----------
// One thread per (b, g, tap, pixel); handles all 18 channels of its group.
__global__ __launch_bounds__(256) void k_sample(
    const float* __restrict__ fu, const float* __restrict__ offs,
    unsigned short* __restrict__ cols, int total) {
  int t = blockIdx.x * 256 + threadIdx.x;
  if (t >= total) return;
  int pix = t & 4095;
  int rem = t >> 12;
  int r = rem % 9;  rem /= 9;
  int g = rem & 7;
  int b = rem >> 3;
  int y = pix >> 6, x = pix & 63;
  int chy = (g * 9 + r) * 2;                 // offset.reshape(B,OFFG,9,2,H,W)
  float oy = offs[(b * M1 + chy) * HW + pix];
  float ox = offs[(b * M1 + chy + 1) * HW + pix];
  float py = (float)(y - 1 + r / 3) + oy;
  float px = (float)(x - 1 + r % 3) + ox;
  float y0f = floorf(py), x0f = floorf(px);
  float fy = py - y0f, fx = px - x0f;
  int y0 = (int)y0f, x0 = (int)x0f;
  float wgt[4];
  int   idx[4];
  bool  val[4];
#pragma unroll
  for (int tap = 0; tap < 4; ++tap) {
    int dy = tap >> 1, dx = tap & 1;
    int yi = y0 + dy, xi = x0 + dx;
    float wy = dy ? fy : (1.f - fy);
    float wx = dx ? fx : (1.f - fx);
    val[tap] = (yi >= 0) & (yi < H_) & (xi >= 0) & (xi < W_);
    idx[tap] = (yi & 63) * W_ + (xi & 63);   // only used when valid
    wgt[tap] = wy * wx;
  }
  const float* base = fu + b * C_ * HW;
  int ncol = b * HW + pix;
  for (int cg = 0; cg < 18; ++cg) {
    int c = g * 18 + cg;
    const float* ch = base + c * HW;
    float a = 0.f;
#pragma unroll
    for (int tap = 0; tap < 4; ++tap)
      if (val[tap]) a += wgt[tap] * ch[idx[tap]];
    cols[(c * 9 + r) * NPIX + ncol] = f2bf(a);
  }
}

// ---------------- K3: main GEMM (WMMA bf16) + bias + SiLU -----------------
__global__ __launch_bounds__(256) void k_conv2_wmma(
    const unsigned short* __restrict__ cols,
    const unsigned short* __restrict__ w2,
    const float* __restrict__ bias,
    float* __restrict__ out) {
  __shared__ __align__(16) unsigned short As[16 * AS_S];
  __shared__ __align__(16) unsigned short Bs[TN * BS_S];
  const int tid  = threadIdx.x;
  const int wave = tid >> 5;
  const int lane = tid & 31;
  const int half = lane >> 4;
  const int l15  = lane & 15;
  const int n0 = blockIdx.x * TN;
  const int m0 = blockIdx.y * 16;
  v8f acc[4] = {{}, {}, {}, {}};
  for (int kc = 0; kc < K2P / 32; ++kc) {
    {  // stage A
      int m = tid >> 4, kp = (tid & 15) * 2;
      *(unsigned int*)&As[m * AS_S + kp] =
          *(const unsigned int*)&w2[(m0 + m) * K2P + kc * 32 + kp];
    }
    // stage B: linear rows of cols; thread owns (n, 8 consecutive k)
#pragma unroll
    for (int i = 0; i < 8; ++i) {
      int idx = tid + i * 256;
      int nl  = idx & 511;
      int kg  = idx >> 9;
      ushort8 vals;
#pragma unroll
      for (int j = 0; j < 8; ++j)
        vals[j] = cols[(kc * 32 + kg * 8 + j) * NPIX + n0 + nl];
      *(ushort8*)&Bs[nl * BS_S + kg * 8] = vals;
    }
    __syncthreads();
    wave_mma4(As, Bs, wave, half, l15, acc);
    __syncthreads();
  }
#pragma unroll
  for (int s = 0; s < 4; ++s) {
    int n = n0 + wave * 64 + s * 16 + l15;
    int b = n >> 12, pix = n & 4095;
#pragma unroll
    for (int r = 0; r < 8; ++r) {
      int m = m0 + r + half * 8;
      float v = acc[s][r] + bias[m];
      out[(b * M2 + m) * HW + pix] = v / (1.f + __expf(-v));   // SiLU
    }
  }
}

// ---------------------------------------------------------------------------
extern "C" void kernel_launch(void* const* d_in, const int* in_sizes, int n_in,
                              void* d_out, int out_size, void* d_ws, size_t ws_size,
                              hipStream_t stream) {
  (void)in_sizes; (void)n_in; (void)out_size; (void)ws_size;
  const float* feat_arm = (const float*)d_in[0];
  const float* feat_up  = (const float*)d_in[1];
  const float* off_w    = (const float*)d_in[2];
  const float* off_b    = (const float*)d_in[3];
  const float* dcn_w    = (const float*)d_in[4];
  const float* dcn_b    = (const float*)d_in[5];
  float* out = (float*)d_out;

  // Workspace carving (256B aligned)
  char* ws = (char*)d_ws;
  size_t p = 0;
  auto carve = [&](size_t bytes) {
    void* r = ws + p;
    p += (bytes + 255) & ~(size_t)255;
    return r;
  };
  unsigned short* xpad = (unsigned short*)carve((size_t)B_ * CIN1 * HP * WP * 2);
  unsigned short* w1bf = (unsigned short*)carve((size_t)M1 * K1D * 2);
  unsigned short* w2bf = (unsigned short*)carve((size_t)M2 * K2P * 2);
  float*          offs = (float*)        carve((size_t)B_ * M1 * HW * 4);
  unsigned short* cols = (unsigned short*)carve((size_t)K2P * NPIX * 2);

  {
    int n = B_ * CIN1 * HP * WP;
    k_pack_xpad<<<(n + 255) / 256, 256, 0, stream>>>(feat_arm, feat_up, xpad, n);
  }
  {
    int n = M1 * K1D;
    k_cvt_w1<<<(n + 255) / 256, 256, 0, stream>>>(off_w, w1bf, n);
  }
  {
    int n = M2 * K2P;
    k_cvt_w2<<<(n + 255) / 256, 256, 0, stream>>>(dcn_w, w2bf, n);
  }
  {
    int n = (K2P - K2D) * NPIX;
    k_zero_colpad<<<(n + 255) / 256, 256, 0, stream>>>(cols, n);
  }
  // K1: offset conv (N -> 128 tiles of 512, M=144 -> 9 tiles of 16)
  {
    dim3 grid(NPIX / TN, M1 / 16);
    k_conv1_wmma<<<grid, 256, 0, stream>>>(xpad, w1bf, off_b, offs);
  }
  // K2: deformable bilinear sampling
  {
    int total = B_ * OFFG * 9 * HW;
    k_sample<<<(total + 255) / 256, 256, 0, stream>>>(feat_up, offs, cols, total);
  }
  // K3: main GEMM + bias + SiLU (M=256 -> 16 tiles)
  {
    dim3 grid(NPIX / TN, M2 / 16);
    k_conv2_wmma<<<grid, 256, 0, stream>>>(cols, w2bf, dcn_b, out);
  }
}
